// FBankPreprocessor1Sec_47897475285227
// MI455X (gfx1250) — matrix operand order repaired and checked by reading, not compiled, not run on hardware
//
#include <hip/hip_runtime.h>
#include <math.h>

typedef __attribute__((ext_vector_type(16))) __bf16 v16bf;
typedef __attribute__((ext_vector_type(8)))  float  v8f;

#define SR        48000
#define WIN       1200
#define HOP       480
#define NFFT      2048
#define NC        1024            // complex length after real-pair packing
#define NMEL      128
#define TFRAMES   128
#define NFRAMES   98              // 1 + (48000-1200)/480
#define KT        33              // K tiles of 32 (1025 bins zero-padded to 1056)
#define SPEC_STRIDE 1064          // ushorts per spec row: 16B aligned, bank-staggered
#define PREEMPH   0.97f
#define MEANV     (-4.2677393f)
#define STDV      (4.5689974f)
#define EPSV      (1.1920928955078125e-07f)
#define PI_F      3.14159265358979323846f
// e^{-i*pi/1024} for the odd half-step in the real-FFT split
#define C_HALF    0.99999529380957619f
#define S_HALF    (-0.00306795676296598f)

__device__ __forceinline__ unsigned short f2bf(float x) {
  unsigned int u = __float_as_uint(x);
  unsigned int r = (u + 0x7FFFu + ((u >> 16) & 1u)) >> 16;   // round-nearest-even
  return (unsigned short)r;
}

__device__ __forceinline__ float melscale(float f) {
  return 1127.0f * logf(1.0f + f * (1.0f / 700.0f));
}

// mel filterbank weight for mel row n, fft bin k. k>=1024 -> 0 (matches the
// reference's appended zero column + our zero K-padding).
__device__ __forceinline__ float fbval(int n, int k) {
  if (k >= 1024) return 0.0f;
  float ml    = melscale(20.0f);
  float mh    = melscale(0.5f * (float)SR);
  float delta = (mh - ml) / (float)(NMEL + 1);
  float left  = ml + (float)n * delta;
  float m     = melscale(((float)SR / (float)NFFT) * (float)k);
  float up    = (m - left) / delta;
  float down  = (left + 2.0f * delta - m) / delta;
  return fmaxf(0.0f, fminf(up, down));
}

// ---------------------------------------------------------------------------
// Kernel 1: fill output with the padding value (covers time frames 98..127)
// ---------------------------------------------------------------------------
__global__ void fill_out_kernel(float* __restrict__ out, int n, float val) {
  int i = blockIdx.x * blockDim.x + threadIdx.x;
  if (i < n) out[i] = val;
}

// ---------------------------------------------------------------------------
// Kernel 2: build WMMA-ready bf16 B-matrix image of the mel filterbank in ws.
// Layout (dwords): Bws[((t*8 + j)*32 + lane)*8 + v]
//   dword packs bf16 pair (K = t*32 + (lane>>4)*8 + 2(v&3) + 16(v>>2), +1), mel col n =
//   j*16 + (lane&15) — the CDNA5 16-bit 32x16 B VGPR layout.
// ---------------------------------------------------------------------------
__global__ void build_b_kernel(unsigned int* __restrict__ Bws) {
  int id = blockIdx.x * blockDim.x + threadIdx.x;
  const int total = KT * 8 * 32 * 8;
  if (id >= total) return;
  int v    = id & 7;
  int lane = (id >> 3) & 31;
  int j    = (id >> 8) & 7;
  int t    = id >> 11;
  int n    = j * 16 + (lane & 15);
  int k    = t * 32 + (lane >> 4) * 8 + (v & 3) * 2 + (v >> 2) * 16;
  unsigned short lo = f2bf(fbval(n, k));
  unsigned short hi = f2bf(fbval(n, k + 1));
  Bws[id] = (unsigned int)lo | ((unsigned int)hi << 16);
}

// ---------------------------------------------------------------------------
// Kernel 3: fused frame-prep + real FFT + bf16 WMMA mel projection.
// One workgroup = (batch, 16-frame tile); 256 threads = 8 waves; each wave
// owns one frame per half-pass. Only 2 block barriers total; all FFT syncs
// are wave-local s_wait_dscnt fences (wave32 lockstep + in-order LDS).
// ---------------------------------------------------------------------------
__global__ void __launch_bounds__(256)
fbank_kernel(const float* __restrict__ wav, const unsigned int* __restrict__ Bws,
             float* __restrict__ out) {
  __shared__ float2 zbuf[8][NC];                                  // 64 KB
  __shared__ __align__(16) unsigned short spec[16][SPEC_STRIDE];  // ~33.3 KB
  __shared__ float2 wtab[NC / 2];                                 // 4 KB twiddles

  const int tid  = threadIdx.x;
  const int lane = tid & 31;
  const int w    = tid >> 5;            // wave id 0..7
  const int tile = blockIdx.x;          // 0..6 frame tiles
  const int b    = blockIdx.y;          // batch
  const int frameBase = tile * 16;

  // one-shot shared init: clear spec (zero K-pad tail), build twiddle table
  {
    unsigned int* sp = (unsigned int*)&spec[0][0];
    for (int i = tid; i < 16 * SPEC_STRIDE / 2; i += 256) sp[i] = 0u;
    for (int i = tid; i < NC / 2; i += 256) {
      float ang = -2.0f * PI_F * (float)i / (float)NC;
      float sn, cs;
      __sincosf(ang, &sn, &cs);
      wtab[i] = make_float2(cs, sn);
    }
  }
  __syncthreads();

  float* zf  = (float*)&zbuf[w][0];     // this wave's frame slot (2048 floats)
  float2* z  = &zbuf[w][0];

  for (int h = 0; h < 2; ++h) {
    const int  g     = frameBase + h * 8 + w;   // global frame index
    const bool valid = (g < NFRAMES);
    const float* src = wav + (size_t)b * SR + (size_t)g * HOP;

    // zero this wave's zero-pad tail (whole slot if frame is out of range)
    {
      float4* zp4 = (float4*)zf;
      const float4 z4 = make_float4(0.f, 0.f, 0.f, 0.f);
      int i0 = (valid ? (WIN / 4) : 0) + lane;  // floats [1200,2048) or all
      for (int i4 = i0; i4 < NC * 2 / 4; i4 += 32) zp4[i4] = z4;
    }

    float mu = 0.0f;
    if (valid) {
      // pass 1: single global read — accumulate mean, stage raw into LDS
      float s = 0.0f;
      for (int i = lane; i < WIN; i += 32) {
        float x = src[i];
        s += x;
        zf[i] = x;
      }
      #pragma unroll
      for (int off = 16; off > 0; off >>= 1) s += __shfl_xor(s, off, 32);
      mu = s * (1.0f / (float)WIN);
    }
    asm volatile("s_wait_dscnt 0" ::: "memory");   // raw staged + tail zeroed

    if (valid) {
      // pass 2: pre-emphasis (x[i-1] via shfl_up + carry) + Hann, in place
      const float c0 = 2.0f * PI_F / (float)(WIN - 1);
      float carry = 0.0f;
      for (int m = 0; m < (WIN + 31) / 32; ++m) {
        int i = lane + 32 * m;
        float x  = (i < WIN) ? zf[i] : 0.0f;
        float xp = __shfl_up(x, 1, 32);
        if (lane == 0) xp = (m == 0) ? x : carry;
        carry = __shfl(x, 31, 32);
        if (i < WIN) {
          float y  = (x - mu) - PREEMPH * (xp - mu);
          float wn = 0.5f - 0.5f * __cosf(c0 * (float)i);
          zf[i] = y * wn;   // store index >= every load index this iter: safe
        }
      }
    }
    asm volatile("s_wait_dscnt 0" ::: "memory");

    // in-place radix-2 DIF FFT (1024-pt complex), LDS twiddle table,
    // output in bit-reversed order
    int s = 0;
    for (int span = NC >> 1; span >= 1; span >>= 1, ++s) {
      for (int m = 0; m < (NC >> 1) / 32; ++m) {
        int i   = lane + 32 * m;
        int off = i & (span - 1);
        int i0  = ((i & ~(span - 1)) << 1) | off;
        int i1  = i0 + span;
        float2 a  = z[i0];
        float2 bb = z[i1];
        float2 tw = wtab[off << s];
        float2 sum = make_float2(a.x + bb.x, a.y + bb.y);
        float2 d   = make_float2(a.x - bb.x, a.y - bb.y);
        z[i0] = sum;
        z[i1] = make_float2(d.x * tw.x - d.y * tw.y,
                            d.x * tw.y + d.y * tw.x);
      }
      asm volatile("s_wait_dscnt 0" ::: "memory"); // wave-local stage fence
    }

    // real-FFT split: X[k] = E + W*O, X[1024-k] = conj(E - W*O); |X|^2 -> bf16
    const int row = h * 8 + w;
    for (int m = 0; m <= 16; ++m) {
      int k = lane + 32 * m;
      if (k <= 512) {
        float2 Zk = z[__brev((unsigned)k) >> 22];
        int   km  = (NC - k) & (NC - 1);
        float2 Zm = z[__brev((unsigned)km) >> 22];
        float Ex = 0.5f * (Zk.x + Zm.x);
        float Ey = 0.5f * (Zk.y - Zm.y);
        float Ax = Zk.x - Zm.x;           // A = Zk - conj(Zm)
        float Ay = Zk.y + Zm.y;
        float Ox = 0.5f * Ay;             // O = A / (2i)
        float Oy = -0.5f * Ax;
        // W = e^{-i*pi*k/1024} = wtab[k>>1] * (odd ? e^{-i*pi/1024} : 1)
        float2 wb = wtab[k >> 1];
        float fc = (k & 1) ? C_HALF : 1.0f;
        float fs = (k & 1) ? S_HALF : 0.0f;
        float Wx = wb.x * fc - wb.y * fs;
        float Wy = wb.x * fs + wb.y * fc;
        float WOx = Ox * Wx - Oy * Wy;
        float WOy = Ox * Wy + Oy * Wx;
        float Xpx = Ex + WOx, Xpy = Ey + WOy;
        float Xmx = Ex - WOx, Xmy = Ey - WOy;
        spec[row][k]            = f2bf(Xpx * Xpx + Xpy * Xpy);
        spec[row][NFFT / 2 - k] = f2bf(Xmx * Xmx + Xmy * Xmy);
      }
    }
    // next half reuses this wave's z slot; per-wave LDS ops are in order
  }
  __syncthreads();   // all 16 spec rows visible to all waves

  // ---- WMMA mel projection: wave j computes 16 frames x 16 mel bins ----
  const int j     = w;
  const int khalf = lane >> 4;
  const int Mrow  = lane & 15;
  union Frag { uint4 q[2]; v16bf v; };
  v8f acc = {0.f, 0.f, 0.f, 0.f, 0.f, 0.f, 0.f, 0.f};

  const uint4* B4 = (const uint4*)Bws;
  for (int t = 0; t < KT; ++t) {
    Frag a, bf;
    int kb = t * 32 + khalf * 8;        // A 16-bit layout: lanes0-15 K0..7/16..23
    a.q[0] = *(const uint4*)&spec[Mrow][kb];
    a.q[1] = *(const uint4*)&spec[Mrow][kb + 16];
    size_t bi = ((size_t)(t * 8 + j) * 32 + (size_t)lane) * 2;
    bf.q[0] = B4[bi];
    bf.q[1] = B4[bi + 1];
    acc = __builtin_amdgcn_wmma_f32_16x16x32_bf16(
        /*neg_a=*/false, a.v, /*neg_b=*/false, bf.v,
        /*c_mod=*/(short)0, acc, /*reuse_a=*/false, /*reuse_b=*/false);
  }

  // epilogue: log, normalize, store (C layout: VGPR r -> M = r + khalf*8)
  const float inv = 1.0f / (2.0f * STDV);
  #pragma unroll
  for (int r = 0; r < 8; ++r) {
    int fr = frameBase + r + khalf * 8;
    if (fr < NFRAMES) {
      float mel = acc[r];
      float val = (logf(fmaxf(mel, EPSV)) - MEANV) * inv;
      out[((size_t)b * TFRAMES + (size_t)fr) * NMEL + (size_t)(j * 16 + Mrow)] = val;
    }
  }
}

// ---------------------------------------------------------------------------
extern "C" void kernel_launch(void* const* d_in, const int* in_sizes, int n_in,
                              void* d_out, int out_size, void* d_ws, size_t ws_size,
                              hipStream_t stream) {
  const float*  wav = (const float*)d_in[0];
  float*        out = (float*)d_out;
  unsigned int* Bws = (unsigned int*)d_ws;   // needs KT*8*32*8*4 = 270,336 B

  const int batches = in_sizes[0] / SR;      // 512
  const float padVal = (0.0f - MEANV) / (STDV * 2.0f);

  fill_out_kernel<<<(out_size + 255) / 256, 256, 0, stream>>>(out, out_size, padVal);

  const int btotal = KT * 8 * 32 * 8;
  build_b_kernel<<<(btotal + 255) / 256, 256, 0, stream>>>(Bws);

  dim3 grid((NFRAMES + 15) / 16, batches);   // 7 x 512 workgroups, 256 threads
  fbank_kernel<<<grid, 256, 0, stream>>>(wav, Bws, out);
}
